// MLA_43370579755051
// MI455X (gfx1250) — matrix operand order repaired and checked by reading, not compiled, not run on hardware
//
#include <hip/hip_runtime.h>

// ---------------- problem constants ----------------
#define B_  2
#define S_  2048
#define D_  2048
#define H_  16
#define DQK 192
#define DN  128
#define DR  64
#define DC  512
#define DV  128
#define BS_ (B_ * S_)
#define SCALE_ 0.0721687836487032206f   // 192^-0.5

// ---------------- WMMA plumbing ----------------
typedef __bf16 v16bf __attribute__((ext_vector_type(16)));
typedef float  v8f   __attribute__((ext_vector_type(8)));

union FragBF {
    v16bf v;
    unsigned int u[8];
};

__device__ __forceinline__ v8f wmma_bf16(const FragBF& a, const FragBF& b, v8f c) {
    return __builtin_amdgcn_wmma_f32_16x16x32_bf16(false, a.v, false, b.v,
                                                   (short)0, c, false, false);
}

__device__ __forceinline__ unsigned short f32_to_bf16(float f) {
    unsigned int u = __float_as_uint(f);
    u += 0x7FFFu + ((u >> 16) & 1u);   // round to nearest even
    return (unsigned short)(u >> 16);
}
__device__ __forceinline__ float bf16_to_f32(unsigned short h) {
    return __uint_as_float(((unsigned int)h) << 16);
}

// Fragment layouts (wave32):
// A (16x32): lane m = lane&15, hi = lane>>4; u[v] packs K = hi*16 + 2v, K+1
// B (32x16): lane n = lane&15, hi = lane>>4; u[v] packs K = hi*16 + 2v, K+1
// C/D (16x16 f32): lane n = lane&15; element r -> row m = r + (lane>>4)*8

// ---------------- elementwise helpers ----------------
__global__ void cvt_f32_bf16(const float* __restrict__ s, unsigned short* __restrict__ d, long n) {
    for (long i = blockIdx.x * (long)blockDim.x + threadIdx.x; i < n;
         i += (long)gridDim.x * blockDim.x)
        d[i] = f32_to_bf16(s[i]);
}

__global__ void prep_wkvb(const float* __restrict__ w,
                          unsigned short* __restrict__ wkT,
                          unsigned short* __restrict__ wbv) {
    const long total = (long)H_ * (DN + DV) * DC;
    for (long i = blockIdx.x * (long)blockDim.x + threadIdx.x; i < total;
         i += (long)gridDim.x * blockDim.x) {
        int c = (int)(i % DC);
        long t = i / DC;
        int row = (int)(t % (DN + DV));
        int h = (int)(t / (DN + DV));
        unsigned short v = f32_to_bf16(w[i]);
        if (row < DN) wkT[((long)h * DC + c) * DN + row] = v;
        else          wbv[((long)h * DV + (row - DN)) * DC + c] = v;
    }
}

__global__ void rope_q(const unsigned short* __restrict__ q,
                       const float* __restrict__ fc, const float* __restrict__ fs,
                       unsigned short* __restrict__ qpe) {
    const long total = (long)BS_ * H_ * (DR / 2);
    for (long i = blockIdx.x * (long)blockDim.x + threadIdx.x; i < total;
         i += (long)gridDim.x * blockDim.x) {
        int p = (int)(i % (DR / 2));
        long t = i / (DR / 2);
        int h = (int)(t % H_);
        long bs = t / H_;
        int s = (int)(bs % S_);
        const unsigned short* src = q + bs * (H_ * DQK) + h * DQK + DN + 2 * p;
        float xr = bf16_to_f32(src[0]);
        float xi = bf16_to_f32(src[1]);
        float c = fc[s * (DR / 2) + p];
        float sn = fs[s * (DR / 2) + p];
        unsigned short* dst = qpe + bs * (H_ * DR) + h * DR + 2 * p;
        dst[0] = f32_to_bf16(xr * c - xi * sn);
        dst[1] = f32_to_bf16(xr * sn + xi * c);
    }
}

__global__ void kv_epilogue(const float* __restrict__ kvfull,
                            const float* __restrict__ kvw,
                            const float* __restrict__ fc, const float* __restrict__ fs,
                            unsigned short* __restrict__ kvc,
                            unsigned short* __restrict__ kvt,
                            unsigned short* __restrict__ pec) {
    const long bs = blockIdx.x;
    const int b = (int)(bs / S_);
    const int s = (int)(bs % S_);
    const float* row = kvfull + bs * (DC + DR);

    __shared__ float red[8];
    __shared__ float s_rn;
    float ss = 0.f;
    for (int c = threadIdx.x; c < DC; c += 256) {
        float v = row[c];
        ss += v * v;
    }
#pragma unroll
    for (int o = 16; o > 0; o >>= 1) ss += __shfl_xor(ss, o, 32);
    if ((threadIdx.x & 31) == 0) red[threadIdx.x >> 5] = ss;
    __syncthreads();
    if (threadIdx.x == 0) {
        float tot = 0.f;
#pragma unroll
        for (int i = 0; i < 8; ++i) tot += red[i];
        s_rn = rsqrtf(tot / (float)DC + 1e-6f);
    }
    __syncthreads();
    const float rn = s_rn;
    for (int c = threadIdx.x; c < DC; c += 256) {
        unsigned short hv = f32_to_bf16(row[c] * rn * kvw[c]);
        kvc[bs * DC + c] = hv;
        kvt[((long)b * DC + c) * S_ + s] = hv;
    }
    if (threadIdx.x < DR / 2) {
        int p = threadIdx.x;
        float xr = row[DC + 2 * p];
        float xi = row[DC + 2 * p + 1];
        float c = fc[s * (DR / 2) + p];
        float sn = fs[s * (DR / 2) + p];
        pec[bs * DR + 2 * p]     = f32_to_bf16(xr * c - xi * sn);
        pec[bs * DR + 2 * p + 1] = f32_to_bf16(xr * sn + xi * c);
    }
}

// ---------------- generic batched NT GEMM: C = A[M,K] * Bw[N,K]^T ----------------
// 4 waves / block arranged 2x2; each wave computes a 32(M) x 64(N) register tile
// (2x4 accumulators -> 8 WMMAs per 6 fragment loads per K-step).
// Block tile = 64(M) x 128(N). M must be a multiple of 64; N guarded.
// Per-batch offsets: off = (z/zdiv)*s1 + (z%zdiv)*s2 (element units).
__global__ __launch_bounds__(128) void gemm_nt_wmma(
    const unsigned short* __restrict__ A,
    const unsigned short* __restrict__ Bw,
    void* __restrict__ Cv,
    int M, int N, int K, int lda, int ldb, int ldc, int out_bf16, int zdiv,
    long a_s1, long a_s2, long b_s1, long b_s2, long c_s1, long c_s2) {
    const int lane = threadIdx.x & 31;
    const int wave = threadIdx.x >> 5;
    const int lm = lane & 15, hi = lane >> 4;
    const int m_base = blockIdx.y * 64 + (wave >> 1) * 32;
    const int n_base = blockIdx.x * 128 + (wave & 1) * 64;
    const int z = blockIdx.z;
    const long aoff = (long)(z / zdiv) * a_s1 + (long)(z % zdiv) * a_s2;
    const long boff = (long)(z / zdiv) * b_s1 + (long)(z % zdiv) * b_s2;
    const long coff = (long)(z / zdiv) * c_s1 + (long)(z % zdiv) * c_s2;

    const unsigned short* Arow[2];
    const unsigned short* Brow[4];
#pragma unroll
    for (int i = 0; i < 2; ++i)
        Arow[i] = A + aoff + (long)(m_base + i * 16 + lm) * lda;
#pragma unroll
    for (int j = 0; j < 4; ++j) {
        int nr = n_base + j * 16 + lm;
        if (nr > N - 1) nr = N - 1;          // clamp loads; stores guarded below
        Brow[j] = Bw + boff + (long)nr * ldb;
    }

    v8f acc[2][4];
    const v8f vzero = {};
#pragma unroll
    for (int i = 0; i < 2; ++i)
#pragma unroll
        for (int j = 0; j < 4; ++j) acc[i][j] = vzero;

    for (int k0 = 0; k0 < K; k0 += 32) {
        __builtin_prefetch(Arow[0] + k0 + 64, 0, 1);
        __builtin_prefetch(Brow[0] + k0 + 64, 0, 1);
        FragBF af[2], bf[4];
#pragma unroll
        for (int v = 0; v < 8; ++v) {
            const int ka = k0 + (hi << 4) + (v << 1);
            const int kb = k0 + (hi << 4) + (v << 1);
#pragma unroll
            for (int i = 0; i < 2; ++i) af[i].u[v] = *(const unsigned int*)(Arow[i] + ka);
#pragma unroll
            for (int j = 0; j < 4; ++j) bf[j].u[v] = *(const unsigned int*)(Brow[j] + kb);
        }
#pragma unroll
        for (int i = 0; i < 2; ++i)
#pragma unroll
            for (int j = 0; j < 4; ++j)
                acc[i][j] = wmma_bf16(af[i], bf[j], acc[i][j]);
    }

#pragma unroll
    for (int i = 0; i < 2; ++i)
#pragma unroll
        for (int j = 0; j < 4; ++j) {
            const int n = n_base + j * 16 + lm;
            if (n >= N) continue;
            if (out_bf16) {
                unsigned short* C = (unsigned short*)Cv;
#pragma unroll
                for (int r = 0; r < 8; ++r) {
                    const int m = m_base + i * 16 + r + (hi << 3);
                    C[coff + (long)m * ldc + n] = f32_to_bf16(acc[i][j][r]);
                }
            } else {
                float* C = (float*)Cv;
#pragma unroll
                for (int r = 0; r < 8; ++r) {
                    const int m = m_base + i * 16 + r + (hi << 3);
                    C[coff + (long)m * ldc + n] = acc[i][j][r];
                }
            }
        }
}

// ---------------- fused flash-style MLA attention (no redundant WMMAs) --------
// grid: (S/64, H, B), 128 threads = 4 independent waves.
// Each wave owns 16 query rows and the FULL 512-col O accumulator (32 x v8f).
// Q fragments (16 qa + 2 qpe chunks) pinned in registers for the whole kernel.
// Per 32-key block and wave: 36 score WMMAs + 32 PV WMMAs, 68 B-frag loads.
__global__ __launch_bounds__(128, 1) void attn_kernel(
    const unsigned short* __restrict__ qabs,  // [B,H,S,DC]
    const unsigned short* __restrict__ qpe,   // [B,S,H,DR]
    const unsigned short* __restrict__ kvc,   // [B,S,DC]
    const unsigned short* __restrict__ kvt,   // [B,DC,S]
    const unsigned short* __restrict__ pec,   // [B,S,DR]
    unsigned short* __restrict__ omid)        // [B,H,S,DC]
{
    __shared__ unsigned short s_p[4][16 * 32];     // per-wave P staging, 4 KB

    const int tid = threadIdx.x;
    const int wave = tid >> 5, lane = tid & 31;
    const int lm = lane & 15, hi = lane >> 4;
    const int row0 = blockIdx.x * 64 + wave * 16;  // this wave's 16 rows
    const int h = blockIdx.y, b = blockIdx.z;

    // ---- pin Q fragments in registers ----
    FragBF qa[16], qp[2];
    {
        const unsigned short* qaRow = qabs + (((long)(b * H_ + h)) * S_ + row0 + lm) * DC;
        const unsigned short* qpRow = qpe + ((long)(b * S_ + row0 + lm)) * (H_ * DR) + h * DR;
#pragma unroll
        for (int kc = 0; kc < 16; ++kc)
#pragma unroll
            for (int v = 0; v < 8; ++v) {
                const int ka = (kc << 5) + (hi << 4) + (v << 1);
                qa[kc].u[v] = *(const unsigned int*)(qaRow + ka);
            }
#pragma unroll
        for (int kc = 0; kc < 2; ++kc)
#pragma unroll
            for (int v = 0; v < 8; ++v) {
                const int ka = (kc << 5) + (hi << 4) + (v << 1);
                qp[kc].u[v] = *(const unsigned int*)(qpRow + ka);
            }
    }

    float mrow[8], lrow[8];
    v8f o[32];
    const v8f vzero = {};
#pragma unroll
    for (int r = 0; r < 8; ++r) { mrow[r] = -1e30f; lrow[r] = 0.f; }
#pragma unroll
    for (int c = 0; c < 32; ++c) o[c] = vzero;

    const int kb_count = (row0 + 16 + 31) >> 5;    // causal bound for THIS wave
    for (int kb = 0; kb < kb_count; ++kb) {
        const int t0 = kb * 32;
        v8f s0 = vzero, s1 = vzero;

        // scores: QKᵀ over DC ...
        const unsigned short* kvr0 = kvc + ((long)b * S_ + t0 + lm) * DC;
        const unsigned short* kvr1 = kvr0 + 16 * DC;
#pragma unroll
        for (int kc = 0; kc < 16; ++kc) {
            FragBF b0f, b1f;
#pragma unroll
            for (int v = 0; v < 8; ++v) {
                const int kbi = (kc << 5) + (hi << 4) + (v << 1);
                b0f.u[v] = *(const unsigned int*)(kvr0 + kbi);
                b1f.u[v] = *(const unsigned int*)(kvr1 + kbi);
            }
            s0 = wmma_bf16(qa[kc], b0f, s0);
            s1 = wmma_bf16(qa[kc], b1f, s1);
        }
        // ... plus rope part over DR
        const unsigned short* per0 = pec + ((long)b * S_ + t0 + lm) * DR;
        const unsigned short* per1 = per0 + 16 * DR;
#pragma unroll
        for (int kc = 0; kc < 2; ++kc) {
            FragBF b0f, b1f;
#pragma unroll
            for (int v = 0; v < 8; ++v) {
                const int kbi = (kc << 5) + (hi << 4) + (v << 1);
                b0f.u[v] = *(const unsigned int*)(per0 + kbi);
                b1f.u[v] = *(const unsigned int*)(per1 + kbi);
            }
            s0 = wmma_bf16(qp[kc], b0f, s0);
            s1 = wmma_bf16(qp[kc], b1f, s1);
        }

        // online softmax (row reductions across the 16-lane halves)
#pragma unroll
        for (int r = 0; r < 8; ++r) {
            const int mg = row0 + r + (hi << 3);
            const int ta = t0 + lm, tb = t0 + 16 + lm;
            float v0 = s0[r] * SCALE_ + (ta <= mg ? 0.f : -1e9f);
            float v1 = s1[r] * SCALE_ + (tb <= mg ? 0.f : -1e9f);
            float tm = fmaxf(v0, v1);
#pragma unroll
            for (int ofs = 8; ofs > 0; ofs >>= 1) tm = fmaxf(tm, __shfl_xor(tm, ofs, 32));
            const float mnew = fmaxf(mrow[r], tm);
            const float alpha = __expf(mrow[r] - mnew);
            const float p0 = __expf(v0 - mnew);
            const float p1 = __expf(v1 - mnew);
            float rs = p0 + p1;
#pragma unroll
            for (int ofs = 8; ofs > 0; ofs >>= 1) rs += __shfl_xor(rs, ofs, 32);
            lrow[r] = lrow[r] * alpha + rs;
            mrow[r] = mnew;
#pragma unroll
            for (int c = 0; c < 32; ++c) o[c][r] *= alpha;
            s_p[wave][(r + (hi << 3)) * 32 + lm]      = f32_to_bf16(p0);
            s_p[wave][(r + (hi << 3)) * 32 + 16 + lm] = f32_to_bf16(p1);
        }
        // per-wave LDS ordering: DS ops are in-order; wait + compiler fence
        asm volatile("s_wait_dscnt 0x0" ::: "memory");

        // P (C-layout) -> A-fragment via LDS, then O += P * V over all 512 cols
        FragBF pa;
#pragma unroll
        for (int v = 0; v < 8; ++v) {
            const int ka = (hi << 4) + (v << 1);
            pa.u[v] = *(const unsigned int*)(&s_p[wave][lm * 32 + ka]);
        }
#pragma unroll
        for (int c = 0; c < 32; ++c) {
            const int cb = c << 4;
            FragBF vb;
#pragma unroll
            for (int v = 0; v < 8; ++v) {
                const int tt = t0 + (hi << 4) + (v << 1);
                vb.u[v] = *(const unsigned int*)(kvt + ((long)b * DC + cb + lm) * S_ + tt);
            }
            o[c] = wmma_bf16(pa, vb, o[c]);
        }
    }

    // normalize + store o_mid (bf16)
#pragma unroll
    for (int r = 0; r < 8; ++r) {
        const float inv = 1.0f / lrow[r];
        const long mg = row0 + r + (hi << 3);
        unsigned short* dst = omid + (((long)(b * H_ + h)) * S_ + mg) * DC + lm;
#pragma unroll
        for (int c = 0; c < 32; ++c) dst[c << 4] = f32_to_bf16(o[c][r] * inv);
    }
}

// ---------------- host-side orchestration ----------------
extern "C" void kernel_launch(void* const* d_in, const int* in_sizes, int n_in,
                              void* d_out, int out_size, void* d_ws, size_t ws_size,
                              hipStream_t stream) {
    const float* x    = (const float*)d_in[0];
    const float* fc   = (const float*)d_in[1];
    const float* fs   = (const float*)d_in[2];
    /* d_in[3] = mask: causal mask computed analytically */
    const float* wq   = (const float*)d_in[4];
    const float* wkva = (const float*)d_in[5];
    const float* kvw  = (const float*)d_in[6];
    const float* wkvb = (const float*)d_in[7];
    const float* wo   = (const float*)d_in[8];
    /* d_in[9] = start_pos == 0 */

    char* ws = (char*)d_ws;
    size_t off = 0;
    auto alloc = [&](size_t bytes) -> void* {
        void* p = ws + off;
        off += (bytes + 255) & ~(size_t)255;
        return p;
    };

    unsigned short* xb    = (unsigned short*)alloc((size_t)BS_ * D_ * 2);
    unsigned short* wqb   = (unsigned short*)alloc((size_t)H_ * DQK * D_ * 2);
    unsigned short* wkvab = (unsigned short*)alloc((size_t)(DC + DR) * D_ * 2);
    unsigned short* wob   = (unsigned short*)alloc((size_t)D_ * H_ * DV * 2);
    unsigned short* wkTb  = (unsigned short*)alloc((size_t)H_ * DC * DN * 2);
    unsigned short* wbvb  = (unsigned short*)alloc((size_t)H_ * DV * DC * 2);
    unsigned short* qb    = (unsigned short*)alloc((size_t)BS_ * H_ * DQK * 2);
    unsigned short* qpeb  = (unsigned short*)alloc((size_t)BS_ * H_ * DR * 2);
    float*          kvfl  = (float*)alloc((size_t)BS_ * (DC + DR) * 4);
    unsigned short* kvcb  = (unsigned short*)alloc((size_t)BS_ * DC * 2);
    unsigned short* kvtb  = (unsigned short*)alloc((size_t)B_ * DC * S_ * 2);
    unsigned short* pecb  = (unsigned short*)alloc((size_t)BS_ * DR * 2);
    unsigned short* qabsb = (unsigned short*)alloc((size_t)B_ * H_ * S_ * DC * 2);
    unsigned short* omidb = (unsigned short*)alloc((size_t)B_ * H_ * S_ * DC * 2);
    unsigned short* ohdb  = (unsigned short*)alloc((size_t)BS_ * H_ * DV * 2);

    cvt_f32_bf16<<<2048, 256, 0, stream>>>(x, xb, (long)BS_ * D_);
    cvt_f32_bf16<<<2048, 256, 0, stream>>>(wq, wqb, (long)H_ * DQK * D_);
    cvt_f32_bf16<<<1024, 256, 0, stream>>>(wkva, wkvab, (long)(DC + DR) * D_);
    cvt_f32_bf16<<<2048, 256, 0, stream>>>(wo, wob, (long)D_ * H_ * DV);
    prep_wkvb<<<2048, 256, 0, stream>>>(wkvb, wkTb, wbvb);

    // q = x @ wq^T  (bf16 out): M=4096, N=3072, K=2048
    {
        dim3 g((H_ * DQK + 127) / 128, BS_ / 64, 1);
        gemm_nt_wmma<<<g, 128, 0, stream>>>(xb, wqb, qb, BS_, H_ * DQK, D_,
                                            D_, D_, H_ * DQK, 1, 1, 0, 0, 0, 0, 0, 0);
    }
    rope_q<<<(BS_ * H_ * (DR / 2)) / 256, 256, 0, stream>>>(qb, fc, fs, qpeb);

    // kv_full = x @ wkv_a^T (f32 out): M=4096, N=576, K=2048
    {
        dim3 g((DC + DR + 127) / 128, BS_ / 64, 1);
        gemm_nt_wmma<<<g, 128, 0, stream>>>(xb, wkvab, kvfl, BS_, DC + DR, D_,
                                            D_, D_, DC + DR, 0, 1, 0, 0, 0, 0, 0, 0);
    }
    kv_epilogue<<<BS_, 256, 0, stream>>>(kvfl, kvw, fc, fs, kvcb, kvtb, pecb);

    // q_abs[b,h] = q_nope[b,:,h,:] @ wkT[h]^T  batched over z=b*H+h: M=2048,N=512,K=128
    {
        dim3 g((DC + 127) / 128, S_ / 64, B_ * H_);
        gemm_nt_wmma<<<g, 128, 0, stream>>>(qb, wkTb, qabsb, S_, DC, DN,
                                            H_ * DQK, DN, DC, 1, H_,
                                            (long)S_ * H_ * DQK, (long)DQK,
                                            0, (long)DC * DN,
                                            (long)H_ * S_ * DC, (long)S_ * DC);
    }

    // fused attention -> o_mid
    {
        dim3 g(S_ / 64, H_, B_);
        attn_kernel<<<g, 128, 0, stream>>>(qabsb, qpeb, kvcb, kvtb, pecb, omidb);
    }

    // o_head[b,:,h,:] = o_mid[b,h] @ wbv[h]^T  batched: M=2048, N=128, K=512
    {
        dim3 g((DV + 127) / 128, S_ / 64, B_ * H_);
        gemm_nt_wmma<<<g, 128, 0, stream>>>(omidb, wbvb, ohdb, S_, DV, DC,
                                            DC, DC, H_ * DV, 1, H_,
                                            (long)H_ * S_ * DC, (long)S_ * DC,
                                            0, (long)DV * DC,
                                            (long)S_ * H_ * DV, (long)DV);
    }

    // out = o_head @ wo^T (f32 out): M=4096, N=2048, K=2048
    {
        dim3 g((D_ + 127) / 128, BS_ / 64, 1);
        gemm_nt_wmma<<<g, 128, 0, stream>>>(ohdb, wob, d_out, BS_, D_, H_ * DV,
                                            H_ * DV, H_ * DV, D_, 0, 1, 0, 0, 0, 0, 0, 0);
    }
}
